// OctreeDeconvGnRelu_65532611002971
// MI455X (gfx1250) — matrix operand order repaired and checked by reading, not compiled, not run on hardware
//
#include <hip/hip_runtime.h>
#include <hip/hip_bf16.h>

typedef __attribute__((ext_vector_type(16))) __bf16 v16bf;
typedef __attribute__((ext_vector_type(8)))  __bf16 v8bf;
typedef __attribute__((ext_vector_type(4)))  __bf16 v4bf;
typedef __attribute__((ext_vector_type(8)))  float  v8f;
typedef __attribute__((ext_vector_type(4)))  float  v4f;

#define N_IN_   131072
#define N_OUT_  1048576
#define C_IN_   256
#define C_OUT_  128
#define KTAPS_  27
#define GROUPS_ 32
#define EPS_    1e-5f
#define LDP_    264   // padded LDS row stride in bf16 elements (kills 512B-stride bank conflicts)

// ---------------------------------------------------------------------------
// 1) zero output accumulator + group-stat scratch
// ---------------------------------------------------------------------------
__global__ void odgr_zero_kernel(float* __restrict__ out, float* __restrict__ ws,
                                 long long n) {
  long long i = (long long)blockIdx.x * blockDim.x + threadIdx.x;
  const long long stride = (long long)gridDim.x * blockDim.x;
  for (; i < n; i += stride) out[i] = 0.0f;
  if (blockIdx.x == 0 && threadIdx.x < 2 * GROUPS_) ws[threadIdx.x] = 0.0f;
}

// ---------------------------------------------------------------------------
// 1b) one-time pack: weight[k][kin][n] f32 -> preW[k][n][kin] bf16 (transposed)
//     kin fastest => coalesced 2B writes; source is L2-resident (3.5 MB)
// ---------------------------------------------------------------------------
__global__ __launch_bounds__(256)
void odgr_pack_weights_kernel(const float* __restrict__ weight,
                              __bf16* __restrict__ preW) {
  const long long i = (long long)blockIdx.x * 256 + threadIdx.x; // < 27*128*256
  const int kin = (int)(i & 255);
  const int n   = (int)((i >> 8) & 127);
  const int k   = (int)(i >> 15);
  preW[i] = (__bf16)weight[((size_t)k * C_IN_ + kin) * C_OUT_ + n];
}

// ---------------------------------------------------------------------------
// 2) per-tap GEMM (bf16 WMMA, f32 accumulate) + scatter-add via f32 atomics
//    block = 256 threads = 8 waves; block owns 128 rows of `data`
//    wave w computes rows [w*16, w*16+16) x all 128 output channels
// ---------------------------------------------------------------------------
__global__ __launch_bounds__(256)
void odgr_deconv_wmma_kernel(const float* __restrict__ data,
                             const float* __restrict__ weight,
                             const __bf16* __restrict__ preW,
                             const long long* __restrict__ neigh,
                             float* __restrict__ out,
                             int use_pre) {
  extern __shared__ __bf16 smem[];
  __bf16* sD = smem;               // [128][LDP_]  data tile  (row-major, K contiguous)
  __bf16* sW = smem + 128 * LDP_;  // [128][LDP_]  weight tile TRANSPOSED: [N][K]

  const int t    = threadIdx.x;
  const int lane = t & 31;
  const int wave = t >> 5;             // 0..7
  const int m0   = blockIdx.x * 128;   // first data row of this tile

  // ---- load + convert data tile: float4 loads, 4 rows per iteration, coalesced
#pragma unroll 4
  for (int i = 0; i < 32; ++i) {
    const int row = i * 4 + (t >> 6);
    const int c4  = (t & 63) * 4;
    const v4f v = *(const v4f*)(data + (size_t)(m0 + row) * C_IN_ + c4);
    *(v4bf*)(sD + row * LDP_ + c4) =
        (v4bf){(__bf16)v.x, (__bf16)v.y, (__bf16)v.z, (__bf16)v.w};
  }
  __syncthreads();

  const int hi   = lane >> 4;              // 0 | 1  (lane half)
  const int lj   = lane & 15;
  const int klo  = hi * 8;                 // A-fragment K phase per ISA layout
  const int rowA = wave * 16 + lj;         // A row (within tile) this lane reads
  const int gdrb = m0 + wave * 16 + 8 * hi;// base global data row for this lane's C rows

  const int nW    = t & 127;               // slow path: output channel handled
  const int koffW = t >> 7;                // 0 | 1

  for (int k = 0; k < KTAPS_; ++k) {
    __syncthreads();   // previous tap's sW reads complete
    if (use_pre) {
      // straight bf16 copy, 16B granules: global b128 -> ds_store_b128
      const __bf16* wp = preW + (size_t)k * (C_OUT_ * C_IN_);
#pragma unroll
      for (int i = 0; i < 16; ++i) {
        const int row   = i * 8 + (t >> 5);
        const int chunk = (t & 31) * 8;
        *(v8bf*)(sW + row * LDP_ + chunk) = *(const v8bf*)(wp + row * C_IN_ + chunk);
      }
    } else {
      // fallback: transpose+convert from f32 weights
      const float* wk = weight + (size_t)k * (C_IN_ * C_OUT_);
      for (int i = 0; i < 128; ++i) {
        const int kin = i * 2 + koffW;
        sW[nW * LDP_ + kin] = (__bf16)wk[kin * C_OUT_ + nW];
      }
    }
    __syncthreads();

    // prefetch next tap's packed weights into cache while we compute
    if (use_pre && (k + 1) < KTAPS_) {
      const char* np = (const char*)(preW + (size_t)(k + 1) * (C_OUT_ * C_IN_));
      __builtin_prefetch(np + t * 256, 0, 1);
      __builtin_prefetch(np + t * 256 + 128, 0, 1);
    }

    v8f c[8];
#pragma unroll
    for (int n = 0; n < 8; ++n) c[n] = (v8f){0, 0, 0, 0, 0, 0, 0, 0};

#pragma unroll
    for (int kk = 0; kk < 8; ++kk) {       // K = 256 in steps of 32
      const int kb = kk * 32;
      // A fragment (16x32 bf16): lane<16 -> K {kb..kb+7, kb+16..kb+23}
      //                          lane>=16 -> K {kb+8..kb+15, kb+24..kb+31}
      v8bf alo = *(const v8bf*)(sD + rowA * LDP_ + kb + klo);
      v8bf ahi = *(const v8bf*)(sD + rowA * LDP_ + kb + 16 + klo);
      v16bf a  = __builtin_shufflevector(alo, ahi,
                   0, 1, 2, 3, 4, 5, 6, 7, 8, 9, 10, 11, 12, 13, 14, 15);
#pragma unroll
      for (int n = 0; n < 8; ++n) {        // 8 N-tiles of 16 cols, A reused
        const int col = n * 16 + lj;       // B column for this lane
        // B fragment (32x16 bf16): lanes<16 hold K {kb..kb+7, kb+16..kb+23}, lanes>=16 {kb+8..kb+15, kb+24..kb+31}
        v8bf blo = *(const v8bf*)(sW + col * LDP_ + kb + klo);
        v8bf bhi = *(const v8bf*)(sW + col * LDP_ + kb + 16 + klo);
        v16bf b  = __builtin_shufflevector(blo, bhi,
                     0, 1, 2, 3, 4, 5, 6, 7, 8, 9, 10, 11, 12, 13, 14, 15);
        c[n] = __builtin_amdgcn_wmma_f32_16x16x32_bf16(
                 false, a, false, b, (short)0, c[n], false, false);
      }
    }

    // ---- scatter-add this tap's 16x128 strip per documented C/D layout:
    //      VGPR j of lane L holds element (M = j + 8*(L>=16), N = L&15 + n*16)
#pragma unroll
    for (int j = 0; j < 8; ++j) {
      const long long orow = neigh[(size_t)(gdrb + j) * KTAPS_ + k];
      float* dst = out + orow * C_OUT_;
#pragma unroll
      for (int n = 0; n < 8; ++n)
        atomicAdd(dst + n * 16 + lj, c[n][j]);
    }
  }
}

// ---------------------------------------------------------------------------
// 3) GroupNorm statistics: sum / sumsq per group (4 channels x N_OUT rows)
// ---------------------------------------------------------------------------
__global__ __launch_bounds__(256)
void odgr_gn_stats_kernel(const float* __restrict__ out, float* __restrict__ ws) {
  __shared__ float ssum[256];
  __shared__ float ssq[256];
  const int t    = threadIdx.x;
  const int c    = t & 127;   // channel
  const int half = t >> 7;    // which row of the pair
  float s = 0.0f, q = 0.0f;
  for (long long r = (long long)blockIdx.x * 2 + half; r < N_OUT_;
       r += (long long)gridDim.x * 2) {
    const float v = out[r * C_OUT_ + c];
    s += v;
    q += v * v;
  }
  ssum[t] = s;
  ssq[t]  = q;
  __syncthreads();
  if (t < GROUPS_) {
    float gs = 0.0f, gq = 0.0f;
#pragma unroll
    for (int i = 0; i < 4; ++i) {
      const int cc = t * 4 + i;
      gs += ssum[cc] + ssum[cc + 128];
      gq += ssq[cc] + ssq[cc + 128];
    }
    atomicAdd(&ws[t * 2 + 0], gs);
    atomicAdd(&ws[t * 2 + 1], gq);
  }
}

// ---------------------------------------------------------------------------
// 4) normalize + affine + ReLU
// ---------------------------------------------------------------------------
__global__ __launch_bounds__(256)
void odgr_gn_apply_kernel(float* __restrict__ out, const float* __restrict__ ws,
                          const float* __restrict__ gamma,
                          const float* __restrict__ beta, long long n) {
  const float cnt = (float)((long long)N_OUT_ * (C_OUT_ / GROUPS_));
  long long i = (long long)blockIdx.x * blockDim.x + threadIdx.x;
  const long long stride = (long long)gridDim.x * blockDim.x;
  for (; i < n; i += stride) {
    const int   c    = (int)(i & (C_OUT_ - 1));
    const int   g    = c >> 2;
    const float mean = ws[g * 2 + 0] / cnt;
    const float var  = ws[g * 2 + 1] / cnt - mean * mean;
    const float xn   = (out[i] - mean) * __frsqrt_rn(var + EPS_);
    const float y    = xn * gamma[c] + beta[c];
    out[i] = y > 0.0f ? y : 0.0f;
  }
}

// ---------------------------------------------------------------------------
extern "C" void kernel_launch(void* const* d_in, const int* in_sizes, int n_in,
                              void* d_out, int out_size, void* d_ws, size_t ws_size,
                              hipStream_t stream) {
  const float*     data   = (const float*)d_in[0];
  const float*     weight = (const float*)d_in[1];
  const float*     gamma  = (const float*)d_in[2];
  const float*     beta   = (const float*)d_in[3];
  const long long* neigh  = (const long long*)d_in[4];  // int64 in reference

  float* out = (float*)d_out;                 // [N_OUT, C_OUT] f32
  float* ws  = (float*)d_ws;                  // [32 groups][sum, sumsq] at offset 0

  // packed bf16 transposed weights live in ws after a 256 B stats area
  __bf16* preW = (__bf16*)((char*)d_ws + 256);
  const size_t pre_bytes = 256 + (size_t)KTAPS_ * C_OUT_ * C_IN_ * sizeof(__bf16);
  const int use_pre = (ws_size >= pre_bytes) ? 1 : 0;

  const long long n_elems = (long long)N_OUT_ * C_OUT_;

  odgr_zero_kernel<<<4096, 256, 0, stream>>>(out, ws, n_elems);

  if (use_pre) {
    const int pack_blocks = (KTAPS_ * C_OUT_ * C_IN_) / 256;  // 3456
    odgr_pack_weights_kernel<<<pack_blocks, 256, 0, stream>>>(weight, preW);
  }

  const size_t lds_bytes = (size_t)2 * 128 * LDP_ * sizeof(__bf16);  // 135168 B
  odgr_deconv_wmma_kernel<<<N_IN_ / 128, 256, lds_bytes, stream>>>(
      data, weight, preW, neigh, out, use_pre);

  odgr_gn_stats_kernel<<<2048, 256, 0, stream>>>(out, ws);

  odgr_gn_apply_kernel<<<4096, 256, 0, stream>>>(out, ws, gamma, beta, n_elems);
}
